// GENet_71880572666568
// MI455X (gfx1250) — compile-verified, matrix operand order checked by reading, not run on hardware
//
#include <hip/hip_runtime.h>
#include <hip/hip_bf16.h>
#include <cstdint>

// ---------------------------------------------------------------------------
// GENet for MI455X (gfx1250): fused gather->MLP->scatter with bf16 WMMA.
//   - v_wmma_f32_16x16x32_bf16 for all big GEMMs (edge MLP dominates: ~168 GFLOP)
//   - edge kernel uses 32-edge M-tiles: 4 WMMAs per K-step from 2 A + 2 B frags
//     (2x B-fragment reuse, ~1.3 GB L2 weight traffic instead of 2.6 GB)
//   - fused tiles avoid materializing [E,2H] intermediates (~GBs of HBM saved)
//   - fp32 accumulation + fp32 atomics for segment sums
// ---------------------------------------------------------------------------

#define IN_DIM    64
#define HIDDEN    128
#define N_NODES   10000
#define N_EDGES   160000
#define N_GRAPHS  32

typedef __bf16 bf16_t;
typedef bf16_t v16bf __attribute__((ext_vector_type(16)));
typedef float  v8f   __attribute__((ext_vector_type(8)));

__device__ __forceinline__ bf16_t f2bf(float f) {
  unsigned int u = __builtin_bit_cast(unsigned int, f);
  u += 0x7FFFu + ((u >> 16) & 1u);           // round-to-nearest-even
  unsigned short s = (unsigned short)(u >> 16);
  return __builtin_bit_cast(bf16_t, s);
}

// Load one 16x32 bf16 operand fragment (A row-major or B as W^T[N][K]).
// CDNA5 layout: lanes 0-15 -> index l, K chunks {k0..k0+7, k0+16..k0+23};
//               lanes 16-31 -> index l-16, K chunks {k0+8..k0+15, k0+24..k0+31}.
__device__ __forceinline__ v16bf load_frag(const bf16_t* __restrict__ base,
                                           int ld, int row, int khalf, int k0) {
  const bf16_t* p = base + row * ld + k0 + khalf;
  v16bf r;
#pragma unroll
  for (int i = 0; i < 8; ++i) { r[i] = p[i]; r[i + 8] = p[i + 16]; }
  return r;
}

__device__ __forceinline__ v8f wmma_bf16(v16bf a, v16bf b, v8f c) {
  return __builtin_amdgcn_wmma_f32_16x16x32_bf16(false, a, false, b,
                                                 (short)0, c, false, false);
}

#define ZERO8 {0.f,0.f,0.f,0.f,0.f,0.f,0.f,0.f}

// ---------------------------------------------------------------------------
// Weight preprocessing: dst[c*R + r] = bf16(src[r*C + c])   (W[K][N] -> Wt[N][K])
// ---------------------------------------------------------------------------
__global__ void w2bf_t_kernel(const float* __restrict__ src, bf16_t* __restrict__ dst,
                              int R, int C) {
  int idx = blockIdx.x * blockDim.x + threadIdx.x;
  if (idx < R * C) {
    int r = idx / C, c = idx % C;
    dst[c * R + r] = f2bf(src[idx]);
  }
}

// ---------------------------------------------------------------------------
// Embedding: h = x @ W_emb + b_emb  (164 MFLOP, scalar is fine), also bf16 copy
// ---------------------------------------------------------------------------
__global__ void embed_kernel(const float* __restrict__ x, const float* __restrict__ W,
                             const float* __restrict__ b, float* __restrict__ h,
                             bf16_t* __restrict__ hbf) {
  int idx = blockIdx.x * blockDim.x + threadIdx.x;
  if (idx >= N_NODES * HIDDEN) return;
  int n = idx >> 7, j = idx & 127;
  float acc = b[j];
  const float* xr = x + n * IN_DIM;
#pragma unroll
  for (int k = 0; k < IN_DIM; ++k) acc += xr[k] * W[k * HIDDEN + j];
  h[idx] = acc;
  hbf[idx] = f2bf(acc);
}

// ---------------------------------------------------------------------------
// Edge MLP (dominant kernel): per block of 32 edges,
//   mi = [h[src] | h[tgt]]  (32x256 bf16, LDS)
//   t  = relu(mi @ W1 + b1) (WMMA, restaged bf16 in LDS)
//   m  = t @ W2 + b2        (WMMA, fp32)
//   agg[tgt] += m           (global fp32 atomics)
// Each wave: 2 M-frags x 2 N-tiles -> 4 WMMAs per K-step, 2x B reuse.
// ---------------------------------------------------------------------------
#define TILE_E 32
#define LDA_E  264
__global__ __launch_bounds__(256) void edge_msg_kernel(
    const bf16_t* __restrict__ hbf, const int* __restrict__ tgt,
    const int* __restrict__ src,
    const bf16_t* __restrict__ W1t, const float* __restrict__ b1,
    const bf16_t* __restrict__ W2t, const float* __restrict__ b2,
    float* __restrict__ agg) {
  __shared__ bf16_t sA[TILE_E * LDA_E];
  __shared__ bf16_t sT[TILE_E * LDA_E];
  __shared__ int sTgt[TILE_E];
  const int tid = threadIdx.x;
  const int e0 = blockIdx.x * TILE_E;
  if (tid < TILE_E) sTgt[tid] = tgt[e0 + tid];
  {  // gather: thread -> 32 contiguous bf16 (64 B) of one mi row
    const int row = tid >> 3;            // 0..31
    const int c0 = (tid & 7) * 32;       // 0..224, chunk stays in one source half
    const int e = e0 + row;
    const int node = (c0 < HIDDEN) ? src[e] : tgt[e];
    const int cc = c0 & (HIDDEN - 1);
    const uint4* p = (const uint4*)(hbf + node * HIDDEN + cc);
    uint4* q = (uint4*)(sA + row * LDA_E + c0);
    q[0] = p[0]; q[1] = p[1]; q[2] = p[2]; q[3] = p[3];
  }
  __syncthreads();

  const int lane  = tid & 31;
  const int wave  = tid >> 5;
  const int m16   = lane & 15;
  const int khalf = (lane >> 4) * 8;
  const int rbase = (lane >> 4) * 8;

  {  // GEMM1: K=256, wave covers output cols [wave*32, wave*32+32), rows 0..31
    v8f acc[2][2] = { { ZERO8, ZERO8 }, { ZERO8, ZERO8 } };
#pragma unroll
    for (int k = 0; k < 8; ++k) {
      v16bf a0 = load_frag(sA, LDA_E, m16, khalf, k * 32);
      v16bf a1 = load_frag(sA + 16 * LDA_E, LDA_E, m16, khalf, k * 32);
#pragma unroll
      for (int nt = 0; nt < 2; ++nt) {
        v16bf b = load_frag(W1t, 256, (wave * 2 + nt) * 16 + m16, khalf, k * 32);
        acc[0][nt] = wmma_bf16(a0, b, acc[0][nt]);
        acc[1][nt] = wmma_bf16(a1, b, acc[1][nt]);
      }
    }
#pragma unroll
    for (int mh = 0; mh < 2; ++mh) {
#pragma unroll
      for (int nt = 0; nt < 2; ++nt) {
        const int col = (wave * 2 + nt) * 16 + m16;
        const float bias = b1[col];
#pragma unroll
        for (int r = 0; r < 8; ++r) {
          float v = acc[mh][nt][r] + bias;
          v = v > 0.f ? v : 0.f;
          sT[(mh * 16 + rbase + r) * LDA_E + col] = f2bf(v);
        }
      }
    }
  }
  __syncthreads();

  {  // GEMM2: K=256, then scatter-add into agg[tgt]
    v8f acc[2][2] = { { ZERO8, ZERO8 }, { ZERO8, ZERO8 } };
#pragma unroll
    for (int k = 0; k < 8; ++k) {
      v16bf a0 = load_frag(sT, LDA_E, m16, khalf, k * 32);
      v16bf a1 = load_frag(sT + 16 * LDA_E, LDA_E, m16, khalf, k * 32);
#pragma unroll
      for (int nt = 0; nt < 2; ++nt) {
        v16bf b = load_frag(W2t, 256, (wave * 2 + nt) * 16 + m16, khalf, k * 32);
        acc[0][nt] = wmma_bf16(a0, b, acc[0][nt]);
        acc[1][nt] = wmma_bf16(a1, b, acc[1][nt]);
      }
    }
#pragma unroll
    for (int mh = 0; mh < 2; ++mh) {
#pragma unroll
      for (int nt = 0; nt < 2; ++nt) {
        const int col = (wave * 2 + nt) * 16 + m16;
        const float bias = b2[col];
#pragma unroll
        for (int r = 0; r < 8; ++r) {
          atomicAdd(&agg[sTgt[mh * 16 + rbase + r] * 256 + col],
                    acc[mh][nt][r] + bias);
        }
      }
    }
  }
}

// ---------------------------------------------------------------------------
// Node update: per block of 16 nodes,
//   ui = [agg | h]  (16x384 bf16), u = relu(ui@uW1+b1)@uW2+b2, h += u
// ---------------------------------------------------------------------------
#define LDA_U 392
#define LDT_U 264
__global__ __launch_bounds__(256) void node_upd_kernel(
    const float* __restrict__ agg, float* __restrict__ h, bf16_t* __restrict__ hbf,
    const bf16_t* __restrict__ W1t, const float* __restrict__ b1,
    const bf16_t* __restrict__ W2t, const float* __restrict__ b2) {
  __shared__ bf16_t sA[16 * LDA_U];
  __shared__ bf16_t sT[16 * LDT_U];
  const int tid = threadIdx.x;
  const int n0 = blockIdx.x * 16;
  {  // gather+convert: 16 rows x 384 cols, 24 elems/thread
    const int row = tid >> 4;
    const int c0 = (tid & 15) * 24;
    const int node = n0 + row;
#pragma unroll
    for (int i = 0; i < 24; ++i) {
      const int c = c0 + i;
      bf16_t v = (c < 256) ? f2bf(agg[node * 256 + c])
                           : hbf[node * HIDDEN + (c - 256)];
      sA[row * LDA_U + c] = v;
    }
  }
  __syncthreads();

  const int lane  = tid & 31;
  const int wave  = tid >> 5;
  const int m16   = lane & 15;
  const int khalf = (lane >> 4) * 8;
  const int rbase = (lane >> 4) * 8;

  {  // GEMM1: K=384, N=256 (2 tiles/wave)
    v8f acc[2] = { ZERO8, ZERO8 };
#pragma unroll
    for (int k = 0; k < 12; ++k) {
      v16bf a = load_frag(sA, LDA_U, m16, khalf, k * 32);
#pragma unroll
      for (int nt = 0; nt < 2; ++nt) {
        v16bf b = load_frag(W1t, 384, (wave * 2 + nt) * 16 + m16, khalf, k * 32);
        acc[nt] = wmma_bf16(a, b, acc[nt]);
      }
    }
#pragma unroll
    for (int nt = 0; nt < 2; ++nt) {
      const int col = (wave * 2 + nt) * 16 + m16;
      const float bias = b1[col];
#pragma unroll
      for (int r = 0; r < 8; ++r) {
        float v = acc[nt][r] + bias;
        v = v > 0.f ? v : 0.f;
        sT[(rbase + r) * LDT_U + col] = f2bf(v);
      }
    }
  }
  __syncthreads();

  {  // GEMM2: K=256, N=128 (1 tile/wave); residual write of h and hbf
    v8f acc = ZERO8;
#pragma unroll
    for (int k = 0; k < 8; ++k) {
      v16bf a = load_frag(sT, LDT_U, m16, khalf, k * 32);
      v16bf b = load_frag(W2t, 256, wave * 16 + m16, khalf, k * 32);
      acc = wmma_bf16(a, b, acc);
    }
    const int col = wave * 16 + m16;
    const float bias = b2[col];
#pragma unroll
    for (int r = 0; r < 8; ++r) {
      const int node = n0 + rbase + r;
      float nv = h[node * HIDDEN + col] + acc[r] + bias;
      h[node * HIDDEN + col] = nv;
      hbf[node * HIDDEN + col] = f2bf(nv);
    }
  }
}

// ---------------------------------------------------------------------------
// Readout: nx = h @ node_W + b; g[batch] += sigmoid(nx[:, :H]) * nx[:, H:]
// ---------------------------------------------------------------------------
#define LDA_R 136
#define LDNX  264
__global__ __launch_bounds__(256) void readout_kernel(
    const bf16_t* __restrict__ hbf, const int* __restrict__ batch,
    const bf16_t* __restrict__ Wt, const float* __restrict__ nb,
    float* __restrict__ g) {
  __shared__ bf16_t sA[16 * LDA_R];
  __shared__ float sNX[16 * LDNX];
  const int tid = threadIdx.x;
  const int n0 = blockIdx.x * 16;
  {
    const int row = tid >> 4;
    const int c0 = (tid & 15) * 8;
    const uint4* p = (const uint4*)(hbf + (n0 + row) * HIDDEN + c0);
    uint4* q = (uint4*)(sA + row * LDA_R + c0);
    q[0] = p[0];
  }
  __syncthreads();

  const int lane  = tid & 31;
  const int wave  = tid >> 5;
  const int m16   = lane & 15;
  const int khalf = (lane >> 4) * 8;
  const int rbase = (lane >> 4) * 8;

  {  // GEMM: K=128, N=256 (2 tiles/wave)
    v8f acc[2] = { ZERO8, ZERO8 };
#pragma unroll
    for (int k = 0; k < 4; ++k) {
      v16bf a = load_frag(sA, LDA_R, m16, khalf, k * 32);
#pragma unroll
      for (int nt = 0; nt < 2; ++nt) {
        v16bf b = load_frag(Wt, 128, (wave * 2 + nt) * 16 + m16, khalf, k * 32);
        acc[nt] = wmma_bf16(a, b, acc[nt]);
      }
    }
#pragma unroll
    for (int nt = 0; nt < 2; ++nt) {
      const int col = (wave * 2 + nt) * 16 + m16;
      const float bias = nb[col];
#pragma unroll
      for (int r = 0; r < 8; ++r)
        sNX[(rbase + r) * LDNX + col] = acc[nt][r] + bias;
    }
  }
  __syncthreads();

  {  // gate * value, pooled into g[batch]
    const int row = tid >> 4;
    const int c0 = (tid & 15) * 8;
    const int b = batch[n0 + row];
#pragma unroll
    for (int i = 0; i < 8; ++i) {
      const int c = c0 + i;
      float gate = sNX[row * LDNX + c];
      gate = 1.f / (1.f + __expf(-gate));
      atomicAdd(&g[b * HIDDEN + c], sNX[row * LDNX + HIDDEN + c] * gate);
    }
  }
}

// ---------------------------------------------------------------------------
// Final: out = g @ graph_W + graph_b  (32x128x128, tiny)
// ---------------------------------------------------------------------------
__global__ void graph_out_kernel(const float* __restrict__ g,
                                 const float* __restrict__ W,
                                 const float* __restrict__ b,
                                 float* __restrict__ out) {
  int idx = blockIdx.x * blockDim.x + threadIdx.x;
  if (idx >= N_GRAPHS * HIDDEN) return;
  int gi = idx >> 7, j = idx & 127;
  float acc = b[j];
#pragma unroll 8
  for (int k = 0; k < HIDDEN; ++k) acc += g[gi * HIDDEN + k] * W[k * HIDDEN + j];
  out[idx] = acc;
}

// ---------------------------------------------------------------------------
extern "C" void kernel_launch(void* const* d_in, const int* in_sizes, int n_in,
                              void* d_out, int out_size, void* d_ws, size_t ws_size,
                              hipStream_t stream) {
  const float* x1      = (const float*)d_in[0];
  const int*   ei1     = (const int*)  d_in[1];
  const int*   batch1  = (const int*)  d_in[2];
  const float* x2      = (const float*)d_in[3];
  const int*   ei2     = (const int*)  d_in[4];
  const int*   batch2  = (const int*)  d_in[5];
  const float* W_emb   = (const float*)d_in[6];
  const float* b_emb   = (const float*)d_in[7];
  const float* msg_W1  = (const float*)d_in[8];
  const float* msg_b1  = (const float*)d_in[9];
  const float* msg_W2  = (const float*)d_in[10];
  const float* msg_b2  = (const float*)d_in[11];
  const float* upd_W1  = (const float*)d_in[12];
  const float* upd_b1  = (const float*)d_in[13];
  const float* upd_W2  = (const float*)d_in[14];
  const float* upd_b2  = (const float*)d_in[15];
  const float* node_W  = (const float*)d_in[16];
  const float* node_b  = (const float*)d_in[17];
  const float* graph_W = (const float*)d_in[18];
  const float* graph_b = (const float*)d_in[19];
  float* out = (float*)d_out;

  uintptr_t base = (uintptr_t)d_ws;
  auto alloc = [&](size_t bytes) -> void* {
    base = (base + 255) & ~(uintptr_t)255;
    void* r = (void*)base; base += bytes; return r;
  };
  float*  h      = (float*) alloc((size_t)N_NODES * 128 * 4);
  bf16_t* hbf    = (bf16_t*)alloc((size_t)N_NODES * 128 * 2);
  float*  agg    = (float*) alloc((size_t)N_NODES * 256 * 4);
  float*  g      = (float*) alloc((size_t)N_GRAPHS * 128 * 4);
  bf16_t* msgW1t = (bf16_t*)alloc((size_t)2 * 256 * 256 * 2);
  bf16_t* msgW2t = (bf16_t*)alloc((size_t)2 * 256 * 256 * 2);
  bf16_t* updW1t = (bf16_t*)alloc((size_t)2 * 384 * 256 * 2);
  bf16_t* updW2t = (bf16_t*)alloc((size_t)2 * 256 * 128 * 2);
  bf16_t* nodeWt = (bf16_t*)alloc((size_t)256 * 128 * 2);

  // Weight transpose + bf16 conversion (deterministic, every call)
  for (int l = 0; l < 2; ++l) {
    w2bf_t_kernel<<<(65536 + 255) / 256, 256, 0, stream>>>(msg_W1 + l * 65536, msgW1t + l * 65536, 256, 256);
    w2bf_t_kernel<<<(65536 + 255) / 256, 256, 0, stream>>>(msg_W2 + l * 65536, msgW2t + l * 65536, 256, 256);
    w2bf_t_kernel<<<(98304 + 255) / 256, 256, 0, stream>>>(upd_W1 + l * 98304, updW1t + l * 98304, 384, 256);
    w2bf_t_kernel<<<(32768 + 255) / 256, 256, 0, stream>>>(upd_W2 + l * 32768, updW2t + l * 32768, 256, 128);
  }
  w2bf_t_kernel<<<(32768 + 255) / 256, 256, 0, stream>>>(node_W, nodeWt, 128, 256);

  for (int br = 0; br < 2; ++br) {
    const float* x     = br ? x2 : x1;
    const int*   ei    = br ? ei2 : ei1;
    const int*   batch = br ? batch2 : batch1;
    const int*   tgtp  = ei;            // edge_index[0]
    const int*   srcp  = ei + N_EDGES;  // edge_index[1]

    embed_kernel<<<(N_NODES * HIDDEN) / 256, 256, 0, stream>>>(x, W_emb, b_emb, h, hbf);

    for (int l = 0; l < 2; ++l) {
      hipMemsetAsync(agg, 0, (size_t)N_NODES * 256 * 4, stream);
      edge_msg_kernel<<<N_EDGES / TILE_E, 256, 0, stream>>>(
          hbf, tgtp, srcp, msgW1t + l * 65536, msg_b1 + l * 256,
          msgW2t + l * 65536, msg_b2 + l * 256, agg);
      node_upd_kernel<<<N_NODES / 16, 256, 0, stream>>>(
          agg, h, hbf, updW1t + l * 98304, upd_b1 + l * 256,
          updW2t + l * 32768, upd_b2 + l * 128);
    }

    hipMemsetAsync(g, 0, (size_t)N_GRAPHS * 128 * 4, stream);
    readout_kernel<<<N_NODES / 16, 256, 0, stream>>>(hbf, batch, nodeWt, node_b, g);
    graph_out_kernel<<<16, 256, 0, stream>>>(g, graph_W, graph_b, out + br * N_GRAPHS * HIDDEN);
  }
}
